// SelfAttention_81458349736091
// MI455X (gfx1250) — matrix operand order repaired and checked by reading, not compiled, not run on hardware
//
#include <hip/hip_runtime.h>

typedef __attribute__((ext_vector_type(16))) __bf16 v16bf;
typedef __attribute__((ext_vector_type(8)))  __bf16 v8bf;
typedef __attribute__((ext_vector_type(4)))  __bf16 v4bf;
typedef __attribute__((ext_vector_type(2)))  __bf16 v2bf;
typedef __attribute__((ext_vector_type(8)))  float  v8f;

#define B_SZ   8
#define S_SZ   2048
#define D_IN   1024
#define D_H    64

// ---------- helpers ----------
__device__ __forceinline__ v16bf pack4f4(const float4& a, const float4& b,
                                         const float4& c, const float4& d) {
    v16bf r;
    r[0]  = (__bf16)a.x; r[1]  = (__bf16)a.y; r[2]  = (__bf16)a.z; r[3]  = (__bf16)a.w;
    r[4]  = (__bf16)b.x; r[5]  = (__bf16)b.y; r[6]  = (__bf16)b.z; r[7]  = (__bf16)b.w;
    r[8]  = (__bf16)c.x; r[9]  = (__bf16)c.y; r[10] = (__bf16)c.z; r[11] = (__bf16)c.w;
    r[12] = (__bf16)d.x; r[13] = (__bf16)d.y; r[14] = (__bf16)d.z; r[15] = (__bf16)d.w;
    return r;
}

__device__ __forceinline__ v16bf cat8(v8bf a, v8bf b) {
    v16bf r;
#pragma unroll
    for (int e = 0; e < 8; ++e) { r[e] = a[e]; r[8 + e] = b[e]; }
    return r;
}

// ---------- kernel 0: one-shot fp32 -> bf16 weight conversion (W is tiny, reused 16384x) ----
__global__ __launch_bounds__(256) void
cvt_w_kernel(const float* __restrict__ Wq, const float* __restrict__ Wk,
             const float* __restrict__ Wv, __bf16* __restrict__ Wqb,
             __bf16* __restrict__ Wkb, __bf16* __restrict__ Wvb) {
    const float* src; __bf16* dst;
    if (blockIdx.y == 0)      { src = Wq; dst = Wqb; }
    else if (blockIdx.y == 1) { src = Wk; dst = Wkb; }
    else                      { src = Wv; dst = Wvb; }
    const int i = (blockIdx.x * 256 + threadIdx.x) * 4;  // 4 elems/thread
    float4 f = *(const float4*)(src + i);
    v4bf r;
    r[0] = (__bf16)f.x; r[1] = (__bf16)f.y; r[2] = (__bf16)f.z; r[3] = (__bf16)f.w;
    *(v4bf*)(dst + i) = r;
}

// ---------- kernel 1: fused bf16 projection  P[m,h] = sum_d X[m,d] * W[h,d] ----------
// grid.x * 8 waves -> one 16-row x 64-col output tile per wave; grid.y selects Q/K/V.
// Q/K stored row-major [B*S][64]; V stored TRANSPOSED [B][64][S] so attention's
// P@V B-operand is a contiguous global read (no LDS transpose needed).
__global__ __launch_bounds__(256) void
proj_bf16_kernel(const float* __restrict__ Xq, const float* __restrict__ Xk,
                 const float* __restrict__ Xv, const __bf16* __restrict__ Wqb,
                 const __bf16* __restrict__ Wkb, const __bf16* __restrict__ Wvb,
                 __bf16* __restrict__ Qo, __bf16* __restrict__ Ko,
                 __bf16* __restrict__ Vt) {
    const float* X; const __bf16* W; __bf16* O; float scale; bool vtr;
    if (blockIdx.y == 0)      { X = Xq; W = Wqb; O = Qo; scale = 0.125f; vtr = false; }
    else if (blockIdx.y == 1) { X = Xk; W = Wkb; O = Ko; scale = 1.0f;   vtr = false; }
    else                      { X = Xv; W = Wvb; O = Vt; scale = 1.0f;   vtr = true;  }

    const int lane = threadIdx.x & 31;
    const int wv   = threadIdx.x >> 5;
    const int m0   = (blockIdx.x * 8 + wv) * 16;
    const int hf   = lane >> 4;      // which 16-lane half
    const int row  = lane & 15;

    v8f acc[4] = {v8f{}, v8f{}, v8f{}, v8f{}};
    const float* ap = X + (size_t)(m0 + row) * D_IN;

    for (int kk = 0; kk < D_IN; kk += 32) {
        // stream-ahead on the 192MB fp32 input; uniform clamp -> SALU select, no branch
        const int pf = (kk + 128 < D_IN) ? kk + 128 : kk;
        __builtin_prefetch(ap + pf, 0, 3);

        // A operand: 16x32 tile of X, converted fp32->bf16 on the fly
        const int kb = kk + hf * 8;
        float4 a0 = *(const float4*)(ap + kb);
        float4 a1 = *(const float4*)(ap + kb + 4);
        float4 a2 = *(const float4*)(ap + kb + 16);
        float4 a3 = *(const float4*)(ap + kb + 20);
        v16bf A = pack4f4(a0, a1, a2, a3);

        // gather ALL four B operands first (one big load clause), then WMMA burst
        const int kbB = kk + hf * 8;
        v16bf Bm[4];
#pragma unroll
        for (int nt = 0; nt < 4; ++nt) {
            const __bf16* wp = W + (size_t)(nt * 16 + row) * D_IN + kbB;
            Bm[nt] = cat8(*(const v8bf*)(wp), *(const v8bf*)(wp + 16));
        }
#pragma unroll
        for (int nt = 0; nt < 4; ++nt)
            acc[nt] = __builtin_amdgcn_wmma_f32_16x16x32_bf16(
                false, A, false, Bm[nt], (short)0, acc[nt], false, false);
    }

    if (!vtr) {
        // C layout: VGPR r -> row r + 8*hf, col = (lane&15) + nt*16
#pragma unroll
        for (int nt = 0; nt < 4; ++nt)
#pragma unroll
            for (int r = 0; r < 8; ++r)
                O[(size_t)(m0 + r + 8 * hf) * D_H + nt * 16 + row] =
                    (__bf16)(acc[nt][r] * scale);
    } else {
        // V transposed: per lane, r=0..7 are 8 CONSECUTIVE seq positions -> b128 store
        const int bat = m0 >> 11;          // m0 / S_SZ
        const int s0  = (m0 & (S_SZ - 1)) + 8 * hf;
#pragma unroll
        for (int nt = 0; nt < 4; ++nt) {
            v8bf pk;
#pragma unroll
            for (int r = 0; r < 8; ++r) pk[r] = (__bf16)acc[nt][r];
            *(v8bf*)(O + ((size_t)bat * D_H + nt * 16 + row) * S_SZ + s0) = pk;
        }
    }
}

// ---------- kernel 2: causal flash attention over bf16 Q/K (row-major) + V^T ----------
#define WAVES 4
#define PT_STRIDE 34
#define WAVE_LDS (16 * PT_STRIDE)

__global__ __launch_bounds__(32 * WAVES) void
flash_attn_kernel(const __bf16* __restrict__ Q, const __bf16* __restrict__ K,
                  const __bf16* __restrict__ Vt, float* __restrict__ out) {
    __shared__ __align__(16) __bf16 smem[WAVES * WAVE_LDS];

    const int lane = threadIdx.x & 31;
    const int wv   = threadIdx.x >> 5;
    __bf16* Pt = smem + wv * WAVE_LDS;        // P tile, row-major: [qrow][key]

    const int hf  = lane >> 4;
    const int row = lane & 15;
    const int b   = blockIdx.y;
    const int q0  = (blockIdx.x * WAVES + wv) * 16;

    const __bf16* Qb  = Q  + (size_t)b * S_SZ * D_H;
    const __bf16* Kb  = K  + (size_t)b * S_SZ * D_H;
    const __bf16* Vtb = Vt + (size_t)b * D_H * S_SZ;   // [feature][seq]

    // Q tile as two A operands (d = 0..31 and 32..63), resident for the whole pass
    v16bf aQ[2];
#pragma unroll
    for (int c = 0; c < 2; ++c) {
        const __bf16* qp = Qb + (size_t)(q0 + row) * D_H + c * 32 + hf * 8;
        aQ[c] = cat8(*(const v8bf*)(qp), *(const v8bf*)(qp + 16));
    }

    float m[8], l[8];
    v8f o[4] = {v8f{}, v8f{}, v8f{}, v8f{}};
#pragma unroll
    for (int r = 0; r < 8; ++r) { m[r] = -3.0e38f; l[r] = 0.0f; }

    const int nblocks = (q0 + 47) >> 5;       // ceil((q0+16)/32) causal key blocks
    for (int kbi = 0; kbi < nblocks; ++kbi) {
        const int kb0 = kbi * 32;

        // prefetch next key block (K rows + V^T runs); uniform clamp, no branch
        const int kpf = (kbi + 1 < nblocks) ? kb0 + 32 : kb0;
        __builtin_prefetch(Kb + (size_t)(kpf + lane) * D_H, 0, 3);
        __builtin_prefetch(Vtb + (size_t)lane * S_SZ + kpf, 0, 3);
        __builtin_prefetch(Vtb + (size_t)(lane + 32) * S_SZ + kpf, 0, 3);

        // ---- scores S = Q K^T (Q pre-scaled by 1/sqrt(D_H)); load all B first ----
        v16bf Bk[4];
#pragma unroll
        for (int nt = 0; nt < 2; ++nt)
#pragma unroll
            for (int c = 0; c < 2; ++c) {
                const __bf16* kp =
                    Kb + (size_t)(kb0 + nt * 16 + row) * D_H + c * 32 + hf * 8;
                Bk[nt * 2 + c] = cat8(*(const v8bf*)(kp), *(const v8bf*)(kp + 16));
            }
        v8f s[2];
#pragma unroll
        for (int nt = 0; nt < 2; ++nt) {
            v8f acc = {};
            acc = __builtin_amdgcn_wmma_f32_16x16x32_bf16(
                false, aQ[0], false, Bk[nt * 2 + 0], (short)0, acc, false, false);
            acc = __builtin_amdgcn_wmma_f32_16x16x32_bf16(
                false, aQ[1], false, Bk[nt * 2 + 1], (short)0, acc, false, false);
            s[nt] = acc;
        }

        // ---- causal mask ----
#pragma unroll
        for (int nt = 0; nt < 2; ++nt) {
            const int key = kb0 + nt * 16 + row;
#pragma unroll
            for (int r = 0; r < 8; ++r)
                if (key > q0 + r + 8 * hf) s[nt][r] = -3.0e38f;
        }

        // ---- online softmax (row lives in one 16-lane half; XOR shuffles stay in-half) ----
        float scl[8];
#pragma unroll
        for (int r = 0; r < 8; ++r) {
            float rm = fmaxf(s[0][r], s[1][r]);
            rm = fmaxf(rm, __shfl_xor(rm, 1));
            rm = fmaxf(rm, __shfl_xor(rm, 2));
            rm = fmaxf(rm, __shfl_xor(rm, 4));
            rm = fmaxf(rm, __shfl_xor(rm, 8));
            const float mn = fmaxf(m[r], rm);
            const float sc = __expf(m[r] - mn);
            m[r] = mn;
            const float p0 = __expf(s[0][r] - mn);
            const float p1 = __expf(s[1][r] - mn);
            s[0][r] = p0; s[1][r] = p1;
            float rs = p0 + p1;
            rs += __shfl_xor(rs, 1);
            rs += __shfl_xor(rs, 2);
            rs += __shfl_xor(rs, 4);
            rs += __shfl_xor(rs, 8);
            l[r] = l[r] * sc + rs;
            scl[r] = sc;
        }
#pragma unroll
        for (int nt = 0; nt < 4; ++nt)
#pragma unroll
            for (int r = 0; r < 8; ++r) o[nt][r] *= scl[r];

        // ---- P: C layout -> bf16 -> LDS (row-major); ds ops are in-order per wave ----
#pragma unroll
        for (int r = 0; r < 8; ++r) {
            Pt[(r + 8 * hf) * PT_STRIDE + row]      = (__bf16)s[0][r];
            Pt[(r + 8 * hf) * PT_STRIDE + 16 + row] = (__bf16)s[1][r];
        }

        // ---- reload P in A layout (k-pairs are contiguous within a P row) ----
        v16bf aP;
#pragma unroll
        for (int j = 0; j < 8; ++j) {
            const int k0 = (j < 4 ? 2 * j : 16 + 2 * (j - 4)) + hf * 8;
            v2bf pp = *(const v2bf*)(&Pt[row * PT_STRIDE + k0]);
            aP[2 * j] = pp[0]; aP[2 * j + 1] = pp[1];
        }

        // ---- O += P @ V : load all four V^T B-operands, then WMMA burst ----
        v16bf Bv[4];
#pragma unroll
        for (int nt = 0; nt < 4; ++nt) {
            const __bf16* vp = Vtb + (size_t)(nt * 16 + row) * S_SZ + kb0 + hf * 8;
            Bv[nt] = cat8(*(const v8bf*)(vp), *(const v8bf*)(vp + 16));
        }
#pragma unroll
        for (int nt = 0; nt < 4; ++nt)
            o[nt] = __builtin_amdgcn_wmma_f32_16x16x32_bf16(
                false, aP, false, Bv[nt], (short)0, o[nt], false, false);
    }

    // ---- normalize + store fp32 output ----
    float* op = out + (size_t)b * S_SZ * D_H;
#pragma unroll
    for (int r = 0; r < 8; ++r) {
        const float inv = 1.0f / l[r];
#pragma unroll
        for (int nt = 0; nt < 4; ++nt)
            op[(size_t)(q0 + r + 8 * hf) * D_H + nt * 16 + row] = o[nt][r] * inv;
    }
}

// ---------- launch ----------
extern "C" void kernel_launch(void* const* d_in, const int* in_sizes, int n_in,
                              void* d_out, int out_size, void* d_ws, size_t ws_size,
                              hipStream_t stream) {
    (void)in_sizes; (void)n_in; (void)out_size; (void)ws_size;
    const float* q  = (const float*)d_in[0];
    const float* k  = (const float*)d_in[1];
    const float* v  = (const float*)d_in[2];
    const float* Wq = (const float*)d_in[3];
    const float* Wk = (const float*)d_in[4];
    const float* Wv = (const float*)d_in[5];
    float* out = (float*)d_out;

    const size_t N  = (size_t)B_SZ * S_SZ * D_H;   // 1M elements per projection
    const size_t NW = (size_t)D_H * D_IN;          // 64K elements per weight
    __bf16* Qb  = (__bf16*)d_ws;
    __bf16* Kb  = Qb + N;
    __bf16* Vtb = Kb + N;          // transposed: [B][64][S]
    __bf16* Wqb = Vtb + N;
    __bf16* Wkb = Wqb + NW;
    __bf16* Wvb = Wkb + NW;

    // 64 blocks x 3: convert weights to bf16 once (4 elems/thread)
    cvt_w_kernel<<<dim3(NW / (256 * 4), 3), 256, 0, stream>>>(Wq, Wk, Wv, Wqb, Wkb, Wvb);

    // 16384 rows / 16 per wave / 8 waves per block = 128 blocks; y selects Q/K/V
    proj_bf16_kernel<<<dim3(128, 3), 256, 0, stream>>>(q, k, v, Wqb, Wkb, Wvb,
                                                       Qb, Kb, Vtb);

    // 128 query tiles per batch / 4 waves per block = 32 blocks; y = batch
    flash_attn_kernel<<<dim3(S_SZ / 16 / WAVES, B_SZ), 32 * WAVES, 0, stream>>>(
        Qb, Kb, Vtb, out);
}
